// VisionTransformer_33904471835050
// MI455X (gfx1250) — matrix operand order repaired and checked by reading, not compiled, not run on hardware
//
#include <hip/hip_runtime.h>
#include <hip/hip_bf16.h>
#include <math.h>

typedef __attribute__((ext_vector_type(16))) _Float16     v16h;
typedef __attribute__((ext_vector_type(8)))  float        v8f;
typedef __attribute__((ext_vector_type(8)))  unsigned int v8u;

#define VIT_B   32
#define VIT_N   197
#define VIT_D   768
#define VIT_L   12
#define VIT_H   12
#define VIT_HD  64
#define VIT_DFF 3072
#define VIT_P   196
#define VIT_M   (VIT_B * VIT_N)   /* 6304 rows of tokens */
#define VIT_MP  (VIT_B * VIT_P)   /* 6272 patch rows     */
#define NCLS    5

// ---------------------------------------------------------------------------
// helpers
// ---------------------------------------------------------------------------
__device__ __forceinline__ float gelu_exact(float x) {
    return 0.5f * x * (1.0f + erff(x * 0.70710678118654752f));
}

// A-fragment K index for 16-bit 16x32 A layout (contiguous, matches B):
// lanes 0-15: VGPR i -> K = 2i ; lanes 16-31: +16
__device__ __forceinline__ int ka_idx(int i, int half) {
    return 2 * i + 16 * half;
}

// ---------------------------------------------------------------------------
// Generic tiled GEMM:  C[M,N] = act(A[M,K] @ B + bias) (+ res)
//   A row-major (lda).  B: if b_nk==0 row-major K x N (ldb), else N x K (ldb).
//   act: 0 = none, 1 = exact GELU.  res: optional residual (ldc), may alias C.
// Block = 256 threads (8 waves).  Tile 64x64; each wave computes a 16x32
// strip (2 WMMA tiles sharing one A fragment).  Requires K%32==0, N%64==0.
// Software-pipelined: next K-slab global loads are issued right after the
// first barrier so they overlap the WMMA phase; single LDS buffer.
// ---------------------------------------------------------------------------
__global__ void gemm_wmma_kernel(const float* __restrict__ A, int lda,
                                 const float* __restrict__ Bw, int ldb, int b_nk,
                                 const float* __restrict__ bias,
                                 const float* __restrict__ res,
                                 float* __restrict__ C, int ldc,
                                 int M, int K, int N, int act) {
    constexpr int ASTR = 34;   // f16 units; even -> 4B-aligned pair reads
    constexpr int BSTR = 34;
    __shared__ _Float16 As[64 * ASTR];   // [row m][k]   (k contiguous)
    __shared__ _Float16 Bs[64 * BSTR];   // [col n][k]   (k contiguous!)

    const int t    = threadIdx.x;
    const int lane = t & 31;
    const int wave = t >> 5;          // 0..7
    const int half = lane >> 4;       // 0/1
    const int mn   = lane & 15;       // m for A, n for B/C
    const int mt   = wave & 3;        // 0..3  (16-row sub-tile)
    const int nt   = wave >> 2;       // 0..1  (32-col strip)

    const int n0 = blockIdx.x * 64;
    const int m0 = blockIdx.y * 64;

    // staging thread mappings
    const int ar  = t >> 2;           // 0..63 : A row in tile
    const int acb = (t & 3) * 8;      // A col base (8 floats)
    const int bkr = t >> 3;           // 0..31 : B k-row      (b_nk==0)
    const int bnb = (t & 7) * 8;      // B n base, 8 cols     (b_nk==0)
    const int bnr = t >> 2;           // 0..63 : B n-row      (b_nk==1)
    const int bkb = (t & 3) * 8;      // B k base, 8 floats   (b_nk==1)

    const bool a_valid = (m0 + ar) < M;
    const float* aptr  = A + (size_t)(m0 + ar) * lda + acb;           // +k0
    const float* bptr0 = Bw + (size_t)bkr * ldb + n0 + bnb;           // +k0*ldb
    const float* bptr1 = Bw + (size_t)(n0 + bnr) * ldb + bkb;         // +k0

    // registered tile loader (software pipeline stage)
    auto load_tile = [&](int k0, float4& av0, float4& av1,
                         float4& bv0, float4& bv1) {
        const float4 z = {0.f, 0.f, 0.f, 0.f};
        av0 = z; av1 = z;
        if (a_valid) {
            const float* ap = aptr + k0;
            av0 = *(const float4*)(ap);
            av1 = *(const float4*)(ap + 4);
            if (k0 + 64 < K) __builtin_prefetch(ap + 64, 0, 1);
        }
        if (!b_nk) {
            const float* bp = bptr0 + (size_t)k0 * ldb;
            bv0 = *(const float4*)(bp);
            bv1 = *(const float4*)(bp + 4);
            if (k0 + 64 < K) __builtin_prefetch(bp + (size_t)64 * ldb, 0, 1);
        } else {
            const float* bp = bptr1 + k0;
            bv0 = *(const float4*)(bp);
            bv1 = *(const float4*)(bp + 4);
            if (k0 + 64 < K) __builtin_prefetch(bp + 64, 0, 1);
        }
    };

    v8f acc0 = {}, acc1 = {};
    float4 av0, av1, bv0, bv1;
    load_tile(0, av0, av1, bv0, bv1);     // prologue

    for (int k0 = 0; k0 < K; k0 += 32) {
        // ---- dump staged registers into LDS (f32 -> f16) ----
        {
            _Float16* dst = &As[ar * ASTR + acb];
            dst[0] = (_Float16)av0.x; dst[1] = (_Float16)av0.y;
            dst[2] = (_Float16)av0.z; dst[3] = (_Float16)av0.w;
            dst[4] = (_Float16)av1.x; dst[5] = (_Float16)av1.y;
            dst[6] = (_Float16)av1.z; dst[7] = (_Float16)av1.w;
        }
        if (!b_nk) {        // transpose while storing -> n-major, k-contiguous
            Bs[(bnb + 0) * BSTR + bkr] = (_Float16)bv0.x;
            Bs[(bnb + 1) * BSTR + bkr] = (_Float16)bv0.y;
            Bs[(bnb + 2) * BSTR + bkr] = (_Float16)bv0.z;
            Bs[(bnb + 3) * BSTR + bkr] = (_Float16)bv0.w;
            Bs[(bnb + 4) * BSTR + bkr] = (_Float16)bv1.x;
            Bs[(bnb + 5) * BSTR + bkr] = (_Float16)bv1.y;
            Bs[(bnb + 6) * BSTR + bkr] = (_Float16)bv1.z;
            Bs[(bnb + 7) * BSTR + bkr] = (_Float16)bv1.w;
        } else {            // already k-contiguous
            _Float16* dst = &Bs[bnr * BSTR + bkb];
            dst[0] = (_Float16)bv0.x; dst[1] = (_Float16)bv0.y;
            dst[2] = (_Float16)bv0.z; dst[3] = (_Float16)bv0.w;
            dst[4] = (_Float16)bv1.x; dst[5] = (_Float16)bv1.y;
            dst[6] = (_Float16)bv1.z; dst[7] = (_Float16)bv1.w;
        }
        __syncthreads();

        // ---- issue next K-slab's global loads (overlap with WMMA phase) ----
        if (k0 + 32 < K) load_tile(k0 + 32, av0, av1, bv0, bv1);

        // ---- fragments from LDS as packed b32 pairs ----
        const int arow = mt * 16 + mn;
        v8u au, bu0, bu1;
#pragma unroll
        for (int i = 0; i < 8; ++i) {
            const int k = ka_idx(i, half);
            au[i] = *(const unsigned*)&As[arow * ASTR + k];
        }
        const int bc0 = (nt * 32 + mn) * BSTR + 16 * half;
        const int bc1 = bc0 + 16 * BSTR;
#pragma unroll
        for (int j = 0; j < 8; ++j) {
            bu0[j] = *(const unsigned*)&Bs[bc0 + 2 * j];
            bu1[j] = *(const unsigned*)&Bs[bc1 + 2 * j];
        }
        const v16h a  = __builtin_bit_cast(v16h, au);
        const v16h b0 = __builtin_bit_cast(v16h, bu0);
        const v16h b1 = __builtin_bit_cast(v16h, bu1);
        acc0 = __builtin_amdgcn_wmma_f32_16x16x32_f16(false, a, false, b0,
                                                      (short)0, acc0, false, false);
        acc1 = __builtin_amdgcn_wmma_f32_16x16x32_f16(false, a, false, b1,
                                                      (short)0, acc1, false, false);
        __syncthreads();
    }

    // ---- epilogue ----
    const int col0 = n0 + nt * 32 + mn;
    const int col1 = col0 + 16;
#pragma unroll
    for (int r = 0; r < 8; ++r) {
        const int row = m0 + mt * 16 + 8 * half + r;
        if (row < M) {
            float v0 = acc0[r], v1 = acc1[r];
            if (bias) { v0 += bias[col0]; v1 += bias[col1]; }
            if (act == 1) { v0 = gelu_exact(v0); v1 = gelu_exact(v1); }
            if (res) {
                v0 += res[(size_t)row * ldc + col0];
                v1 += res[(size_t)row * ldc + col1];
            }
            C[(size_t)row * ldc + col0] = v0;
            C[(size_t)row * ldc + col1] = v1;
        }
    }
}

// ---------------------------------------------------------------------------
// Attention scores: S[b,h,i,j] = scale * <q_i, k_j>   (one wave per 16x16 tile)
// grid (ceil(N/16), ceil(N/16), B*H), block 32.
// ---------------------------------------------------------------------------
__global__ void attn_scores_kernel(const float* __restrict__ qkv,
                                   float* __restrict__ S, float scale) {
    const int j0 = blockIdx.x * 16;
    const int i0 = blockIdx.y * 16;
    const int bh = blockIdx.z;
    const int b  = bh / VIT_H;
    const int hh = bh - b * VIT_H;
    const int lane = threadIdx.x & 31;
    const int half = lane >> 4;
    const int mn   = lane & 15;

    const float* qb = qkv + (size_t)(b * VIT_N) * (3 * VIT_D) + hh * VIT_HD;
    const float* kb = qb + VIT_D;

    v8f acc = {};
#pragma unroll
    for (int kk = 0; kk < VIT_HD; kk += 32) {
        v16h a = {}, bm = {};
        const int row  = i0 + mn;
        const bool rv  = row < VIT_N;
        const int col  = j0 + mn;
        const bool cv  = col < VIT_N;
#pragma unroll
        for (int i = 0; i < 8; ++i) {
            const int k = kk + ka_idx(i, half);
            if (rv) {
                float2 t = *(const float2*)(qb + (size_t)row * (3 * VIT_D) + k);
                a[2 * i] = (_Float16)t.x; a[2 * i + 1] = (_Float16)t.y;
            }
        }
#pragma unroll
        for (int j = 0; j < 8; ++j) {
            const int k = kk + 2 * j + 16 * half;
            if (cv) {
                float2 t = *(const float2*)(kb + (size_t)col * (3 * VIT_D) + k);
                bm[2 * j] = (_Float16)t.x; bm[2 * j + 1] = (_Float16)t.y;
            }
        }
        acc = __builtin_amdgcn_wmma_f32_16x16x32_f16(false, a, false, bm,
                                                     (short)0, acc, false, false);
    }
    const int col = j0 + mn;
#pragma unroll
    for (int r = 0; r < 8; ++r) {
        const int row = i0 + 8 * half + r;
        if (row < VIT_N && col < VIT_N)
            S[((size_t)bh * VIT_N + row) * VIT_N + col] = acc[r] * scale;
    }
}

// ---------------------------------------------------------------------------
// O = softmax(S) @ V   (one wave per 16(rows) x 16(d-cols) tile)
// grid (HD/16, ceil(N/16), B*H), block 32.
// ---------------------------------------------------------------------------
__global__ void attn_av_kernel(const float* __restrict__ qkv,
                               const float* __restrict__ S,
                               float* __restrict__ O) {
    const int d0 = blockIdx.x * 16;
    const int i0 = blockIdx.y * 16;
    const int bh = blockIdx.z;
    const int b  = bh / VIT_H;
    const int hh = bh - b * VIT_H;
    const int lane = threadIdx.x & 31;
    const int half = lane >> 4;
    const int mn   = lane & 15;

    const float* Sb = S + (size_t)bh * VIT_N * VIT_N;
    const float* vb = qkv + (size_t)(b * VIT_N) * (3 * VIT_D) + 2 * VIT_D + hh * VIT_HD;

    v8f acc = {};
    for (int j0 = 0; j0 < VIT_N; j0 += 32) {
        v16h a = {}, bm = {};
        const int row = i0 + mn;
        const bool rv = row < VIT_N;
#pragma unroll
        for (int i = 0; i < 8; ++i) {
            const int k = j0 + ka_idx(i, half);
            a[2 * i]     = (rv && k     < VIT_N) ? (_Float16)Sb[(size_t)row * VIT_N + k]     : (_Float16)0.f;
            a[2 * i + 1] = (rv && k + 1 < VIT_N) ? (_Float16)Sb[(size_t)row * VIT_N + k + 1] : (_Float16)0.f;
        }
#pragma unroll
        for (int j = 0; j < 8; ++j) {
            const int kr = j0 + 2 * j + 16 * half;
            bm[2 * j]     = (kr     < VIT_N) ? (_Float16)vb[(size_t)kr       * (3 * VIT_D) + d0 + mn] : (_Float16)0.f;
            bm[2 * j + 1] = (kr + 1 < VIT_N) ? (_Float16)vb[(size_t)(kr + 1) * (3 * VIT_D) + d0 + mn] : (_Float16)0.f;
        }
        acc = __builtin_amdgcn_wmma_f32_16x16x32_f16(false, a, false, bm,
                                                     (short)0, acc, false, false);
    }
#pragma unroll
    for (int r = 0; r < 8; ++r) {
        const int row = i0 + 8 * half + r;
        if (row < VIT_N)
            O[((size_t)b * VIT_N + row) * VIT_D + hh * VIT_HD + d0 + mn] = acc[r];
    }
}

// ---------------------------------------------------------------------------
// LayerNorm over D=768.  grid = rows, block = 256.  in_stride selects rows.
// ---------------------------------------------------------------------------
__global__ void ln_kernel(const float* __restrict__ in,
                          const float* __restrict__ g,
                          const float* __restrict__ bta,
                          float* __restrict__ out, long in_stride) {
    const int row = blockIdx.x;
    const int tid = threadIdx.x;
    const float* p = in + (size_t)row * in_stride;
    float s = 0.f, ss = 0.f;
    for (int k = tid; k < VIT_D; k += 256) { float v = p[k]; s += v; ss += v * v; }
    __shared__ float r1[256], r2[256];
    r1[tid] = s; r2[tid] = ss; __syncthreads();
    for (int st = 128; st > 0; st >>= 1) {
        if (tid < st) { r1[tid] += r1[tid + st]; r2[tid] += r2[tid + st]; }
        __syncthreads();
    }
    const float mean = r1[0] * (1.0f / VIT_D);
    const float var  = r2[0] * (1.0f / VIT_D) - mean * mean;
    const float inv  = rsqrtf(var + 1e-5f);
    float* q = out + (size_t)row * VIT_D;
    for (int k = tid; k < VIT_D; k += 256)
        q[k] = (p[k] - mean) * inv * g[k] + bta[k];
}

// ---------------------------------------------------------------------------
// Row softmax over 197 elements.  grid = B*H*N, block = 64.
// ---------------------------------------------------------------------------
__global__ void softmax_kernel(float* __restrict__ S) {
    const int tid = threadIdx.x;
    float* row = S + (size_t)blockIdx.x * VIT_N;
    __shared__ float red[64];
    float m = -3.0e38f;
    for (int j = tid; j < VIT_N; j += 64) m = fmaxf(m, row[j]);
    red[tid] = m; __syncthreads();
    for (int st = 32; st > 0; st >>= 1) {
        if (tid < st) red[tid] = fmaxf(red[tid], red[tid + st]);
        __syncthreads();
    }
    m = red[0]; __syncthreads();
    float sum = 0.f;
    for (int j = tid; j < VIT_N; j += 64) {
        float e = __expf(row[j] - m);
        row[j] = e; sum += e;
    }
    red[tid] = sum; __syncthreads();
    for (int st = 32; st > 0; st >>= 1) {
        if (tid < st) red[tid] += red[tid + st];
        __syncthreads();
    }
    const float inv = 1.0f / red[0];
    for (int j = tid; j < VIT_N; j += 64) row[j] *= inv;
}

// ---------------------------------------------------------------------------
// im2col for the 16x16/stride16 patch conv.  out: (B*196) x 768 row-major.
// ---------------------------------------------------------------------------
__global__ void im2col_kernel(const float* __restrict__ x, float* __restrict__ out) {
    const size_t idx = (size_t)blockIdx.x * blockDim.x + threadIdx.x;
    if (idx >= (size_t)VIT_MP * VIT_D) return;
    const int k   = (int)(idx % VIT_D);
    const int row = (int)(idx / VIT_D);
    const int b = row / VIT_P, p = row % VIT_P;
    const int c = k / 256, r = k % 256, ph = r / 16, pw = r % 16;
    const int hy = (p / 14) * 16 + ph;
    const int hx = (p % 14) * 16 + pw;
    out[idx] = x[(((size_t)b * 3 + c) * 224 + hy) * 224 + hx];
}

// ---------------------------------------------------------------------------
// Assemble token sequence: cls + patches, add positional embedding.
// ---------------------------------------------------------------------------
__global__ void assemble_kernel(const float* __restrict__ pe_out,
                                const float* __restrict__ cls_token,
                                const float* __restrict__ pos_embed,
                                float* __restrict__ tok) {
    const size_t idx = (size_t)blockIdx.x * blockDim.x + threadIdx.x;
    if (idx >= (size_t)VIT_M * VIT_D) return;
    const int d   = (int)(idx % VIT_D);
    const int row = (int)(idx / VIT_D);
    const int b = row / VIT_N, n = row % VIT_N;
    float v = (n == 0) ? cls_token[d]
                       : pe_out[((size_t)b * VIT_P + (n - 1)) * VIT_D + d];
    tok[idx] = v + pos_embed[(size_t)n * VIT_D + d];
}

// ---------------------------------------------------------------------------
// Classifier head: out[b, c] = cls[b] . head_w[:, c] + head_b[c]
// grid = B, block = 256.
// ---------------------------------------------------------------------------
__global__ void head_kernel(const float* __restrict__ cls,
                            const float* __restrict__ hw,
                            const float* __restrict__ hb,
                            float* __restrict__ out) {
    const int b   = blockIdx.x;
    const int tid = threadIdx.x;
    __shared__ float red[256];
    for (int c = 0; c < NCLS; ++c) {
        float s = 0.f;
        for (int k = tid; k < VIT_D; k += 256)
            s += cls[(size_t)b * VIT_D + k] * hw[(size_t)k * NCLS + c];
        red[tid] = s; __syncthreads();
        for (int st = 128; st > 0; st >>= 1) {
            if (tid < st) red[tid] += red[tid + st];
            __syncthreads();
        }
        if (tid == 0) out[b * NCLS + c] = red[0] + hb[c];
        __syncthreads();
    }
}

// ---------------------------------------------------------------------------
// host-side orchestration
// ---------------------------------------------------------------------------
static void launch_gemm(const float* A, int lda, const float* B, int ldb, int b_nk,
                        const float* bias, const float* res, float* C, int ldc,
                        int M, int K, int N, int act, hipStream_t stream) {
    dim3 grid(N / 64, (M + 63) / 64);
    gemm_wmma_kernel<<<grid, 256, 0, stream>>>(A, lda, B, ldb, b_nk, bias, res,
                                               C, ldc, M, K, N, act);
}

extern "C" void kernel_launch(void* const* d_in, const int* in_sizes, int n_in,
                              void* d_out, int out_size, void* d_ws, size_t ws_size,
                              hipStream_t stream) {
    (void)in_sizes; (void)n_in; (void)out_size; (void)ws_size;
    const float* x         = (const float*)d_in[0];
    const float* conv_w    = (const float*)d_in[1];   // (768, 768)  N x K
    const float* conv_b    = (const float*)d_in[2];
    const float* cls_token = (const float*)d_in[3];
    const float* pos_embed = (const float*)d_in[4];
    const float* ln1_g     = (const float*)d_in[5];
    const float* ln1_b     = (const float*)d_in[6];
    const float* qkv_w     = (const float*)d_in[7];   // (L, 768, 2304) K x N
    const float* qkv_b     = (const float*)d_in[8];
    const float* proj_w    = (const float*)d_in[9];
    const float* proj_b    = (const float*)d_in[10];
    const float* ln2_g     = (const float*)d_in[11];
    const float* ln2_b     = (const float*)d_in[12];
    const float* fc1_w     = (const float*)d_in[13];
    const float* fc1_b     = (const float*)d_in[14];
    const float* fc2_w     = (const float*)d_in[15];
    const float* fc2_b     = (const float*)d_in[16];
    const float* norm_g    = (const float*)d_in[17];
    const float* norm_b    = (const float*)d_in[18];
    const float* head_w    = (const float*)d_in[19];
    const float* head_b    = (const float*)d_in[20];
    float* out = (float*)d_out;

    // workspace layout (floats)
    float* base = (float*)d_ws;
    size_t off = 0;
    float* tok  = base + off; off += (size_t)VIT_M * VIT_D;        // 6304 x 768
    float* y    = base + off; off += (size_t)VIT_M * VIT_D;        // LN out / cls
    float* qkv  = base + off; off += (size_t)VIT_M * 3 * VIT_D;    // 6304 x 2304
    float* attn = base + off; off += (size_t)VIT_B * VIT_H * VIT_N * VIT_N;
    float* obuf = base + off; off += (size_t)VIT_M * VIT_D;
    float* ffn  = base + off; off += (size_t)VIT_M * VIT_DFF;      // also im2col
    float* pe_cols = ffn;   // (6272 x 768), used before any FFN
    float* pe_out  = qkv;   // (6272 x 768), used before any QKV

    const int tpb = 256;

    // 1) patch embedding: im2col + GEMM against conv_w^T (+bias)
    {
        const size_t total = (size_t)VIT_MP * VIT_D;
        im2col_kernel<<<(unsigned)((total + tpb - 1) / tpb), tpb, 0, stream>>>(x, pe_cols);
        launch_gemm(pe_cols, VIT_D, conv_w, VIT_D, /*b_nk=*/1, conv_b, nullptr,
                    pe_out, VIT_D, VIT_MP, VIT_D, VIT_D, 0, stream);
        const size_t tot2 = (size_t)VIT_M * VIT_D;
        assemble_kernel<<<(unsigned)((tot2 + tpb - 1) / tpb), tpb, 0, stream>>>(
            pe_out, cls_token, pos_embed, tok);
    }

    // 2) transformer blocks
    for (int l = 0; l < VIT_L; ++l) {
        const float* l1g = ln1_g + (size_t)l * VIT_D;
        const float* l1b = ln1_b + (size_t)l * VIT_D;
        const float* qw  = qkv_w + (size_t)l * VIT_D * 3 * VIT_D;
        const float* qb  = qkv_b + (size_t)l * 3 * VIT_D;
        const float* pw  = proj_w + (size_t)l * VIT_D * VIT_D;
        const float* pb  = proj_b + (size_t)l * VIT_D;
        const float* l2g = ln2_g + (size_t)l * VIT_D;
        const float* l2b = ln2_b + (size_t)l * VIT_D;
        const float* f1w = fc1_w + (size_t)l * VIT_D * VIT_DFF;
        const float* f1b = fc1_b + (size_t)l * VIT_DFF;
        const float* f2w = fc2_w + (size_t)l * VIT_DFF * VIT_D;
        const float* f2b = fc2_b + (size_t)l * VIT_D;

        ln_kernel<<<VIT_M, 256, 0, stream>>>(tok, l1g, l1b, y, VIT_D);
        launch_gemm(y, VIT_D, qw, 3 * VIT_D, 0, qb, nullptr,
                    qkv, 3 * VIT_D, VIT_M, VIT_D, 3 * VIT_D, 0, stream);

        {
            dim3 gs((VIT_N + 15) / 16, (VIT_N + 15) / 16, VIT_B * VIT_H);
            attn_scores_kernel<<<gs, 32, 0, stream>>>(qkv, attn, 0.125f);
            softmax_kernel<<<VIT_B * VIT_H * VIT_N, 64, 0, stream>>>(attn);
            dim3 ga(VIT_HD / 16, (VIT_N + 15) / 16, VIT_B * VIT_H);
            attn_av_kernel<<<ga, 32, 0, stream>>>(qkv, attn, obuf);
        }

        launch_gemm(obuf, VIT_D, pw, VIT_D, 0, pb, /*res=*/tok,
                    tok, VIT_D, VIT_M, VIT_D, VIT_D, 0, stream);

        ln_kernel<<<VIT_M, 256, 0, stream>>>(tok, l2g, l2b, y, VIT_D);
        launch_gemm(y, VIT_D, f1w, VIT_DFF, 0, f1b, nullptr,
                    ffn, VIT_DFF, VIT_M, VIT_D, VIT_DFF, /*gelu=*/1, stream);
        launch_gemm(ffn, VIT_DFF, f2w, VIT_D, 0, f2b, /*res=*/tok,
                    tok, VIT_D, VIT_M, VIT_DFF, VIT_D, 0, stream);
    }

    // 3) final LN on cls rows only, then head
    ln_kernel<<<VIT_B, 256, 0, stream>>>(tok, norm_g, norm_b, y,
                                         (long)VIT_N * VIT_D);
    head_kernel<<<VIT_B, 256, 0, stream>>>(y, head_w, head_b, out);
}
